// GATLayer_55113020342351
// MI455X (gfx1250) — matrix operand order repaired and checked by reading, not compiled, not run on hardware
//
#include <hip/hip_runtime.h>
#include <hip/hip_bf16.h>
#include <math.h>

#define GAT_N      40000
#define GAT_E      640000
#define GAT_IN     256
#define GAT_H      4
#define GAT_C      32
#define GAT_F      128               // HEADS * C
#define GAT_ETOT   (GAT_E + GAT_N)   // edges + self loops
#define NEG_SLOPE  0.2f

typedef __attribute__((ext_vector_type(16))) _Float16 v16h;
typedef __attribute__((ext_vector_type(8)))  _Float16 v8h;
typedef __attribute__((ext_vector_type(8)))  float    v8f;

// ---------------------------------------------------------------------------
// Monotone order-preserving float <-> uint key (for atomicMax-based segment max)
// ---------------------------------------------------------------------------
__device__ __forceinline__ unsigned gat_enc(float f) {
    unsigned u = __float_as_uint(f);
    return (u & 0x80000000u) ? ~u : (u | 0x80000000u);
}
__device__ __forceinline__ float gat_dec(unsigned k) {
    return (k & 0x80000000u) ? __uint_as_float(k & 0x7FFFFFFFu)
                             : __uint_as_float(~k);
}

// ---------------------------------------------------------------------------
// 0) zero output accumulator + softmax state (runs every launch)
// ---------------------------------------------------------------------------
__global__ void gat_init_kernel(float* __restrict__ out,
                                unsigned* __restrict__ emax,
                                float* __restrict__ esum) {
    int i = blockIdx.x * blockDim.x + threadIdx.x;
    if (i < GAT_N * GAT_F) out[i] = 0.0f;
    if (i < GAT_N * GAT_H) { emax[i] = 0u; esum[i] = 0.0f; }
}

// ---------------------------------------------------------------------------
// 0b) one-time per launch: Wt[c][k] = (f16) W[k][c]   (128 x 256 f16, 64 KB)
// ---------------------------------------------------------------------------
__global__ void gat_convert_W_kernel(const float* __restrict__ W,
                                     _Float16* __restrict__ Wt) {
    int i = blockIdx.x * blockDim.x + threadIdx.x;   // i = k*128 + c
    if (i >= GAT_IN * GAT_F) return;
    int k = i >> 7;
    int c = i & (GAT_F - 1);
    Wt[(size_t)c * GAT_IN + k] = (_Float16)W[i];
}

// ---------------------------------------------------------------------------
// 1) xp = x @ W  via V_WMMA_F32_16X16X32_F16 (wave32)
//    block = 8 waves; block stages its 16x256 A tile (f16) in LDS once,
//    wave w owns output cols [16w, 16w+16). B comes from pre-converted
//    transposed f16 W (two 8-half runs per lane per k-step).
// ---------------------------------------------------------------------------
__global__ void __launch_bounds__(256)
gat_gemm_wmma_kernel(const float* __restrict__ x,
                     const _Float16* __restrict__ Wt,
                     float* __restrict__ xp) {
    __shared__ _Float16 sA[16 * GAT_IN];   // 16 rows x 256 K, 8 KB

    const int tid   = threadIdx.x;
    const int mbase = blockIdx.x * 16;

    // stage + convert A tile: each j covers one row, fully coalesced
    #pragma unroll
    for (int j = 0; j < 16; ++j)
        sA[j * GAT_IN + tid] = (_Float16)x[(size_t)(mbase + j) * GAT_IN + tid];
    __syncthreads();

    const int lane  = tid & 31;
    const int wave  = tid >> 5;
    const int lo    = lane & 15;      // row (A) / col (B,D)
    const int hi    = lane >> 4;      // K-half selector
    const int nbase = wave * 16;
    const int col   = nbase + lo;

    const _Float16* wrow = Wt + (size_t)col * GAT_IN;   // this lane's B column
    const _Float16* arow = sA + lo * GAT_IN;            // this lane's A row

    v8f acc = {};
    #pragma unroll
    for (int kb = 0; kb < GAT_IN; kb += 32) {
        // A fragment: halfs [kb+hi*8, +8) and [kb+16+hi*8, +8)  (two b128)
        v8h a0 = *(const v8h*)(arow + kb + hi * 8);
        v8h a1 = *(const v8h*)(arow + kb + 16 + hi * 8);
        v16h a = __builtin_shufflevector(a0, a1,
                 0, 1, 2, 3, 4, 5, 6, 7, 8, 9, 10, 11, 12, 13, 14, 15);

        // B fragment: halfs [kb+hi*8, +8) and [kb+16+hi*8, +8)  (two b128)
        v8h b0 = *(const v8h*)(wrow + kb + hi * 8);
        v8h b1 = *(const v8h*)(wrow + kb + 16 + hi * 8);
        v16h b = __builtin_shufflevector(b0, b1,
                 0, 1, 2, 3, 4, 5, 6, 7, 8, 9, 10, 11, 12, 13, 14, 15);

        acc = __builtin_amdgcn_wmma_f32_16x16x32_f16(
            /*neg_a=*/false, a, /*neg_b=*/false, b,
            /*c_mod=*/(short)0, acc, /*reuse_a=*/false, /*reuse_b=*/false);
    }

    // D: VGPR v, lanes 0-15 -> M=v, lanes 16-31 -> M=v+8, N=lane&15
    #pragma unroll
    for (int v = 0; v < 8; ++v)
        xp[(size_t)(mbase + v + 8 * hi) * GAT_F + nbase + lo] = acc[v];
}

// ---------------------------------------------------------------------------
// 2) per-node attention coefficients: a_src[n,h], a_dst[n,h]
// ---------------------------------------------------------------------------
__global__ void gat_attn_coef_kernel(const float* __restrict__ xp,
                                     const float* __restrict__ att_src,
                                     const float* __restrict__ att_dst,
                                     float* __restrict__ a_src,
                                     float* __restrict__ a_dst) {
    int i = blockIdx.x * blockDim.x + threadIdx.x;   // i = n*H + h
    if (i >= GAT_N * GAT_H) return;
    int h = i & (GAT_H - 1);
    int n = i >> 2;
    const float* xr = xp + (size_t)n * GAT_F + h * GAT_C;
    float ss = 0.0f, sd = 0.0f;
    #pragma unroll
    for (int c = 0; c < GAT_C; ++c) {
        float v = xr[c];
        ss += v * att_src[h * GAT_C + c];
        sd += v * att_dst[h * GAT_C + c];
    }
    a_src[i] = ss;
    a_dst[i] = sd;
}

// ---------------------------------------------------------------------------
// 3) segment max of leaky-relu'd logits over destination nodes
// ---------------------------------------------------------------------------
__global__ void gat_edge_max_kernel(const int* __restrict__ ei,
                                    const float* __restrict__ a_src,
                                    const float* __restrict__ a_dst,
                                    unsigned* __restrict__ emax) {
    int e = blockIdx.x * blockDim.x + threadIdx.x;
    if (e >= GAT_ETOT) return;
    int s, d;
    if (e < GAT_E) { s = ei[e]; d = ei[GAT_E + e]; }
    else           { s = d = e - GAT_E; }            // self loop
    #pragma unroll
    for (int h = 0; h < GAT_H; ++h) {
        float v = a_src[s * GAT_H + h] + a_dst[d * GAT_H + h];
        v = v > 0.0f ? v : NEG_SLOPE * v;
        atomicMax(&emax[d * GAT_H + h], gat_enc(v));
    }
}

// ---------------------------------------------------------------------------
// 4) segment sum of exp(e - max) over destination nodes
// ---------------------------------------------------------------------------
__global__ void gat_edge_sum_kernel(const int* __restrict__ ei,
                                    const float* __restrict__ a_src,
                                    const float* __restrict__ a_dst,
                                    const unsigned* __restrict__ emax,
                                    float* __restrict__ esum) {
    int e = blockIdx.x * blockDim.x + threadIdx.x;
    if (e >= GAT_ETOT) return;
    int s, d;
    if (e < GAT_E) { s = ei[e]; d = ei[GAT_E + e]; }
    else           { s = d = e - GAT_E; }
    #pragma unroll
    for (int h = 0; h < GAT_H; ++h) {
        float v = a_src[s * GAT_H + h] + a_dst[d * GAT_H + h];
        v = v > 0.0f ? v : NEG_SLOPE * v;
        float m = gat_dec(emax[d * GAT_H + h]);
        atomicAdd(&esum[d * GAT_H + h], expf(v - m));
    }
}

// ---------------------------------------------------------------------------
// 5) message aggregation: one wave32 per edge, lane -> 4 contiguous channels
// ---------------------------------------------------------------------------
__global__ void __launch_bounds__(256)
gat_edge_aggregate_kernel(const int* __restrict__ ei,
                          const float* __restrict__ a_src,
                          const float* __restrict__ a_dst,
                          const unsigned* __restrict__ emax,
                          const float* __restrict__ esum,
                          const float* __restrict__ xp,
                          float* __restrict__ out) {
    int gid  = blockIdx.x * blockDim.x + threadIdx.x;
    int e    = gid >> 5;
    int lane = gid & 31;
    if (e >= GAT_ETOT) return;
    int s, d;
    if (e < GAT_E) { s = ei[e]; d = ei[GAT_E + e]; }
    else           { s = d = e - GAT_E; }

    const int c4 = lane * 4;          // channel base [0,128)
    const int h  = lane >> 3;         // 32 channels per head -> 8 lanes per head

    const float* xrow = xp + (size_t)s * GAT_F + c4;
    __builtin_prefetch(xrow, 0, 1);   // gfx1250 global_prefetch_b8

    float v = a_src[s * GAT_H + h] + a_dst[d * GAT_H + h];
    v = v > 0.0f ? v : NEG_SLOPE * v;
    float m     = gat_dec(emax[d * GAT_H + h]);
    float alpha = expf(v - m) / esum[d * GAT_H + h];

    const float4 xv = *(const float4*)xrow;
    float* o = out + (size_t)d * GAT_F + c4;
    atomicAdd(o + 0, alpha * xv.x);
    atomicAdd(o + 1, alpha * xv.y);
    atomicAdd(o + 2, alpha * xv.z);
    atomicAdd(o + 3, alpha * xv.w);
}

// ---------------------------------------------------------------------------
// 6) out = relu(out + bias)
// ---------------------------------------------------------------------------
__global__ void gat_finalize_kernel(float* __restrict__ out,
                                    const float* __restrict__ bias) {
    int i = blockIdx.x * blockDim.x + threadIdx.x;
    if (i >= GAT_N * GAT_F) return;
    float v = out[i] + bias[i & (GAT_F - 1)];
    out[i] = v > 0.0f ? v : 0.0f;
}

// ---------------------------------------------------------------------------
extern "C" void kernel_launch(void* const* d_in, const int* in_sizes, int n_in,
                              void* d_out, int out_size, void* d_ws, size_t ws_size,
                              hipStream_t stream) {
    (void)in_sizes; (void)n_in; (void)out_size; (void)ws_size;

    const float* x       = (const float*)d_in[0];   // [N, 256]
    const int*   ei      = (const int*)d_in[1];     // [2, E]
    /* d_in[2] = batch (unused) */
    const float* W       = (const float*)d_in[3];   // [256, 128]
    const float* att_src = (const float*)d_in[4];   // [4, 32]
    const float* att_dst = (const float*)d_in[5];   // [4, 32]
    const float* bias    = (const float*)d_in[6];   // [128]
    float*       out     = (float*)d_out;           // [N, 128]

    // workspace layout (Wt slice lands 32-byte aligned: 23,040,000 % 32 == 0)
    float*    xp    = (float*)d_ws;                               // N*128 f32
    float*    a_src = xp    + (size_t)GAT_N * GAT_F;              // N*4 f32
    float*    a_dst = a_src + (size_t)GAT_N * GAT_H;              // N*4 f32
    unsigned* emax  = (unsigned*)(a_dst + (size_t)GAT_N * GAT_H); // N*4 u32
    float*    esum  = (float*)(emax + (size_t)GAT_N * GAT_H);     // N*4 f32
    _Float16* Wt    = (_Float16*)(esum + (size_t)GAT_N * GAT_H);  // 128*256 f16

    gat_init_kernel<<<(GAT_N * GAT_F + 255) / 256, 256, 0, stream>>>(out, emax, esum);

    gat_convert_W_kernel<<<(GAT_IN * GAT_F + 255) / 256, 256, 0, stream>>>(W, Wt);

    gat_gemm_wmma_kernel<<<GAT_N / 16, 256, 0, stream>>>(x, Wt, xp);

    gat_attn_coef_kernel<<<(GAT_N * GAT_H + 255) / 256, 256, 0, stream>>>(
        xp, att_src, att_dst, a_src, a_dst);

    gat_edge_max_kernel<<<(GAT_ETOT + 255) / 256, 256, 0, stream>>>(
        ei, a_src, a_dst, emax);

    gat_edge_sum_kernel<<<(GAT_ETOT + 255) / 256, 256, 0, stream>>>(
        ei, a_src, a_dst, emax, esum);

    gat_edge_aggregate_kernel<<<((size_t)GAT_ETOT * 32 + 255) / 256, 256, 0, stream>>>(
        ei, a_src, a_dst, emax, esum, xp, out);

    gat_finalize_kernel<<<(GAT_N * GAT_F + 255) / 256, 256, 0, stream>>>(out, bias);
}
